// TransformerModel_84585085928095
// MI455X (gfx1250) — compile-verified, matrix operand order checked by reading, not run on hardware
//
#include <hip/hip_runtime.h>
#include <math.h>

// ---------------- types ----------------
typedef __bf16 bf16_t;
typedef bf16_t v8bf  __attribute__((ext_vector_type(8)));
typedef bf16_t v16bf __attribute__((ext_vector_type(16)));
typedef float  v8f   __attribute__((ext_vector_type(8)));

#define BB   256
#define SS   120
#define MTOT (BB*SS)      // 30720
#define DM   256
#define FF   2048
#define MC   3840         // M chunk (30720/8), divisible by 256
#define EMBW 1376

__device__ __forceinline__ bf16_t to_bf16(float f) {
  unsigned u = __builtin_bit_cast(unsigned, f);
  unsigned r = u + 0x7FFFu + ((u >> 16) & 1u);   // round-to-nearest-even
  unsigned short h = (unsigned short)(r >> 16);
  return __builtin_bit_cast(bf16_t, h);
}
__device__ __forceinline__ float bf2f(bf16_t b) {
  unsigned short h = __builtin_bit_cast(unsigned short, b);
  unsigned u = ((unsigned)h) << 16;
  return __builtin_bit_cast(float, u);
}

// ---------------- WMMA fragment load ----------------
// 16-bit A-matrix 16x32 layout (ISA 7.12.2): lane L<16 -> row L, K 0..7 & 16..23;
// lane L>=16 -> row L-16, K 8..15 & 24..31.  B is stored transposed [N][K] so the
// same loader serves both operands.
__device__ __forceinline__ v16bf load_frag(const bf16_t* __restrict__ base, int ld,
                                           int row0, int k0, int lane) {
  int r  = row0 + (lane & 15);
  int kk = k0 + ((lane >> 4) << 3);
  const bf16_t* p = base + (size_t)r * ld + kk;
  v8bf lo = *(const v8bf*)p;
  v8bf hi = *(const v8bf*)(p + 16);
  v16bf f;
#pragma unroll
  for (int i = 0; i < 8; ++i) { f[i] = lo[i]; f[i + 8] = hi[i]; }
  return f;
}

// C/D 16x16 f32 layout: VGPR r, lanes 0-15 -> M=r, N=lane; lanes 16-31 -> M=8+r.
__device__ __forceinline__ void store_tile(v8f acc, int rowBase, int colBase, int lane,
                                           const float* __restrict__ bias,
                                           const float* resid, int ldr,
                                           float* Cf, int ldc,
                                           bf16_t* Cb, int ldcb,
                                           int N, int flags) {
  int n = colBase + (lane & 15);
  if (n >= N) return;
  int mb = rowBase + ((lane >> 4) << 3);
  float bv = bias ? bias[n] : 0.f;
#pragma unroll
  for (int r = 0; r < 8; ++r) {
    int m = mb + r;
    float v = acc[r] + bv;
    if (resid) v += resid[(size_t)m * ldr + n];
    if (flags & 1) v = 0.5f * v * (1.f + erff(v * 0.70710678118654752f)); // exact GELU
    if (Cf) Cf[(size_t)m * ldc + n] = v;
    if (Cb) Cb[(size_t)m * ldcb + n] = to_bf16(v);
  }
}

// ---------------- GEMM: C = A(bf16)[M,K] * Bt(bf16)[N,K]^T + bias (+resid) ----------------
// block = 256 threads = 8 waves (4 in M x 2 in N); wave tile 64x64 -> 16 wmma / k-step
// (8 fragment loads feed 16 WMMAs: ~33 MAC/byte from cache vs ~16 with 32x32 tiles).
__global__ void __launch_bounds__(256) gemm_bf16_kernel(
    const bf16_t* __restrict__ A, int lda,
    const bf16_t* __restrict__ Bt, int ldb,
    const float* __restrict__ bias,
    const float* resid, int ldr,
    float* Cf, int ldc,
    bf16_t* Cb, int ldcb,
    int M, int N, int K, int flags) {
  int lane = threadIdx.x & 31;
  int w    = threadIdx.x >> 5;
  int wm   = w & 3, wn = w >> 2;
  int rowBase = blockIdx.x * 256 + wm * 64;
  int colBase = blockIdx.y * 128 + wn * 64;
  v8f acc[4][4];
#pragma unroll
  for (int i = 0; i < 4; ++i)
#pragma unroll
    for (int j = 0; j < 4; ++j) acc[i][j] = v8f{};

  for (int k0 = 0; k0 < K; k0 += 32) {
    v16bf af[4], bfr[4];
#pragma unroll
    for (int i = 0; i < 4; ++i) af[i]  = load_frag(A,  lda, rowBase + 16 * i, k0, lane);
#pragma unroll
    for (int j = 0; j < 4; ++j) bfr[j] = load_frag(Bt, ldb, colBase + 16 * j, k0, lane);
#pragma unroll
    for (int i = 0; i < 4; ++i)
#pragma unroll
      for (int j = 0; j < 4; ++j)
        acc[i][j] = __builtin_amdgcn_wmma_f32_16x16x32_bf16(
            false, af[i], false, bfr[j], (short)0, acc[i][j], false, false);
  }
#pragma unroll
  for (int i = 0; i < 4; ++i)
#pragma unroll
    for (int j = 0; j < 4; ++j)
      store_tile(acc[i][j], rowBase + 16 * i, colBase + 16 * j, lane,
                 bias, resid, ldr, Cf, ldc, Cb, ldcb, N, flags);
}

// ---------------- weight convert+transpose: W[K][N] f32 -> Wt[Npad][K] bf16 ----------------
__global__ void wtrans_kernel(const float* __restrict__ W, int K, int N, int Npad,
                              bf16_t* __restrict__ Wt) {
  long long idx = (long long)blockIdx.x * blockDim.x + threadIdx.x;
  long long total = (long long)Npad * K;
  if (idx >= total) return;
  int n = (int)(idx / K);
  int k = (int)(idx % K);
  float v = (n < N) ? W[(size_t)k * N + n] : 0.f;
  Wt[idx] = to_bf16(v);
}

// ---------------- embedding gather (chunked): out[Mc,1376] bf16 ----------------
__global__ void embed_kernel(const int* __restrict__ x, int m0, int Mc,
                             const float* __restrict__ e0, const float* __restrict__ e1,
                             const float* __restrict__ e2, const float* __restrict__ e3,
                             const float* __restrict__ e4, const float* __restrict__ e5,
                             const float* __restrict__ e6,
                             bf16_t* __restrict__ out) {
  long long idx = (long long)blockIdx.x * blockDim.x + threadIdx.x;
  if (idx >= (long long)Mc * EMBW) return;
  int m = (int)(idx / EMBW), c = (int)(idx % EMBW);
  int gm = m0 + m;
  int i = (c < 128) ? 0 : (c < 384) ? 1 : (c < 448) ? 2 : (c < 480) ? 3 : (c < 992) ? 4 : (c < 1120) ? 5 : 6;
  int off = (i == 0) ? 0 : (i == 1) ? 128 : (i == 2) ? 384 : (i == 3) ? 448 : (i == 4) ? 480 : (i == 5) ? 992 : 1120;
  int tok = x[(size_t)gm * 7 + i];
  int loc = c - off;
  float v;
  switch (i) {
    case 0: v = e0[(size_t)tok * 128 + loc] * 11.3137085f; break;
    case 1: v = e1[(size_t)tok * 256 + loc] * 16.0f;       break;
    case 2: v = e2[(size_t)tok * 64  + loc] * 8.0f;        break;
    case 3: v = e3[(size_t)tok * 32  + loc] * 5.65685425f; break;
    case 4: v = e4[(size_t)tok * 512 + loc] * 22.627417f;  break;
    case 5: v = e5[(size_t)tok * 128 + loc] * 11.3137085f; break;
    default:v = e6[(size_t)tok * 128 + loc] * 11.3137085f; break;
  }
  out[idx] = to_bf16(v);
}

// ---------------- h += pe[s]; hb = bf16(h) ----------------
__global__ void add_pe_kernel(float* __restrict__ h, bf16_t* __restrict__ hb,
                              const float* __restrict__ pe) {
  long long idx = (long long)blockIdx.x * blockDim.x + threadIdx.x;
  if (idx >= (long long)MTOT * DM) return;
  int m = (int)(idx / DM), n = (int)(idx % DM);
  int s = m % SS;
  float v = h[idx] + pe[(size_t)s * DM + n];
  h[idx] = v;
  hb[idx] = to_bf16(v);
}

// ---------------- LayerNorm (row=256), in-place f32 + bf16 copy ----------------
__global__ void __launch_bounds__(256) ln_kernel(float* __restrict__ h, bf16_t* hb,
                                                 const float* __restrict__ g,
                                                 const float* __restrict__ b) {
  int m = blockIdx.x, t = threadIdx.x;
  __shared__ float red[256];
  float v = h[(size_t)m * DM + t];
  red[t] = v; __syncthreads();
  for (int s = 128; s > 0; s >>= 1) { if (t < s) red[t] += red[t + s]; __syncthreads(); }
  float mean = red[0] / 256.f; __syncthreads();
  float d = v - mean;
  red[t] = d * d; __syncthreads();
  for (int s = 128; s > 0; s >>= 1) { if (t < s) red[t] += red[t + s]; __syncthreads(); }
  float var = red[0] / 256.f;
  float y = d * rsqrtf(var + 1e-5f) * g[t] + b[t];
  h[(size_t)m * DM + t] = y;
  if (hb) hb[(size_t)m * DM + t] = to_bf16(y);
}

// ---------------- causal linear attention scan: one block per (b,head) ----------------
// thread e owns state column S[:,e] (64 f32 in VGPRs); q,k,z broadcast via LDS.
__global__ void __launch_bounds__(64) attn_scan_kernel(const bf16_t* __restrict__ qb,
                                                       const bf16_t* __restrict__ kb,
                                                       const bf16_t* __restrict__ vb,
                                                       bf16_t* __restrict__ ab) {
  int bh = blockIdx.x;
  int b = bh >> 2, hh = bh & 3;
  int e = threadIdx.x;
  __shared__ float qs[64], ks[64], zs[64];
  float Scol[64];
#pragma unroll
  for (int d = 0; d < 64; ++d) Scol[d] = 0.f;
  zs[e] = 0.f;
  __syncthreads();
  size_t base = (size_t)b * SS * DM + hh * 64;
  for (int s = 0; s < SS; ++s) {
    size_t off = base + (size_t)s * DM;
    float qv = bf2f(qb[off + e]);
    float kv = bf2f(kb[off + e]);
    float vv = bf2f(vb[off + e]);
    qv = (qv > 0.f ? qv : expf(qv) - 1.f) + 1.f;   // phi = elu+1
    kv = (kv > 0.f ? kv : expf(kv) - 1.f) + 1.f;
    qs[e] = qv; ks[e] = kv; zs[e] += kv;
    __syncthreads();
    float num = 0.f, den = 0.f;
#pragma unroll 8
    for (int d = 0; d < 64; ++d) {
      float kd = ks[d], qd = qs[d];
      Scol[d] += kd * vv;
      num += qd * Scol[d];
      den += qd * zs[d];
    }
    ab[off + e] = to_bf16(num / (den + 1e-6f));
    __syncthreads();
  }
}

// ---------------- build A1 = [hb | tf_skip] bf16 [MTOT,288] ----------------
__global__ void build_a1_kernel(const bf16_t* __restrict__ hb, const int* __restrict__ target,
                                const float* __restrict__ emb3, bf16_t* __restrict__ A1) {
  long long idx = (long long)blockIdx.x * blockDim.x + threadIdx.x;
  if (idx >= (long long)MTOT * 288) return;
  int m = (int)(idx / 288), c = (int)(idx % 288);
  if (c < 256) {
    A1[idx] = hb[(size_t)m * DM + c];
  } else {
    int tok = target[(size_t)m * 7 + 3];
    A1[idx] = to_bf16(emb3[(size_t)tok * 32 + (c - 256)] * 5.65685425f);
  }
}

// ---------------- fill cat[:,256:288] with label embedding ----------------
__global__ void build_label_kernel(const int* __restrict__ label, const float* __restrict__ embL,
                                   bf16_t* __restrict__ catb) {
  long long idx = (long long)blockIdx.x * blockDim.x + threadIdx.x;
  if (idx >= (long long)MTOT * 32) return;
  int m = (int)(idx / 32), j = (int)(idx % 32);
  int b = m / SS;
  catb[(size_t)m * 288 + 256 + j] = to_bf16(embL[(size_t)label[b] * 32 + j] * 5.65685425f);
}

// ---------------- classifier: out[b,5] = y_[b,:] (30720) @ Wcls + bcls ----------------
__global__ void __launch_bounds__(256) cls_kernel(const float* __restrict__ yf,
                                                  const float* __restrict__ Wcls,
                                                  const float* __restrict__ bcls,
                                                  float* __restrict__ out) {
  int b = blockIdx.x, t = threadIdx.x;
  float acc[5] = {0.f, 0.f, 0.f, 0.f, 0.f};
  const float* row = yf + (size_t)b * MTOT;
  for (int k = t; k < MTOT; k += 256) {
    float v = row[k];
#pragma unroll
    for (int j = 0; j < 5; ++j) acc[j] += v * Wcls[(size_t)k * 5 + j];
  }
  __shared__ float red[256];
  for (int j = 0; j < 5; ++j) {
    red[t] = acc[j]; __syncthreads();
    for (int s = 128; s > 0; s >>= 1) { if (t < s) red[t] += red[t + s]; __syncthreads(); }
    if (t == 0) out[(size_t)b * 5 + j] = red[0] + bcls[j];
    __syncthreads();
  }
}

// ======================= host =======================
extern "C" void kernel_launch(void* const* d_in, const int* in_sizes, int n_in,
                              void* d_out, int out_size, void* d_ws, size_t ws_size,
                              hipStream_t stream) {
  (void)in_sizes; (void)n_in; (void)out_size; (void)ws_size;
  const int* x      = (const int*)d_in[0];
  const int* target = (const int*)d_in[1];
  const int* label  = (const int*)d_in[2];
  int ip = 3;
  auto F = [&]() -> const float* { return (const float*)d_in[ip++]; };
  // params leaves in sorted-key flatten order
  const float* Wcat = F();
  const float* Wcls = F();
  const float* Win  = F();
  const float* Wp_barbeat = F(); const float* Wp_chord = F(); const float* Wp_duration = F();
  const float* Wp_pitch = F();   const float* Wp_tempo = F(); const float* Wp_velocity = F();
  const float* Wtype = F();
  const float* bcat = F(); const float* bcls = F(); const float* bfin = F(); const float* bin = F();
  const float* bp_barbeat = F(); const float* bp_chord = F(); const float* bp_duration = F();
  const float* bp_pitch = F();   const float* bp_tempo = F(); const float* bp_velocity = F();
  const float* btype = F();
  const float* emb[7]; for (int i = 0; i < 7; ++i) emb[i] = F();
  const float* emb_label = F();
  const float* gf = F();
  struct Layer { const float *W1,*W2,*Wk,*Wo,*Wq,*Wv,*b1,*b2,*bk,*bo,*bq,*bv,*c1,*c2,*g1,*g2; } L[6];
  for (int l = 0; l < 6; ++l) {
    L[l].W1=F(); L[l].W2=F(); L[l].Wk=F(); L[l].Wo=F(); L[l].Wq=F(); L[l].Wv=F();
    L[l].b1=F(); L[l].b2=F(); L[l].bk=F(); L[l].bo=F(); L[l].bq=F(); L[l].bv=F();
    L[l].c1=F(); L[l].c2=F(); L[l].g1=F(); L[l].g2=F();
  }
  const float* pe = F();

  // ---- workspace allocator ----
  char* ws = (char*)d_ws;
  size_t cur = 0;
  auto alloc = [&](size_t bytes) -> char* {
    char* p = ws + cur;
    cur = (cur + bytes + 255) & ~(size_t)255;
    return p;
  };
  // weight arena (bf16, transposed [Npad][K]); Npad multiple of 128 (block N-tile)
  auto wt = [&](const float* W, int K, int N, int Npad) -> bf16_t* {
    bf16_t* dst = (bf16_t*)alloc((size_t)Npad * K * sizeof(bf16_t));
    long long total = (long long)Npad * K;
    int blocks = (int)((total + 255) / 256);
    wtrans_kernel<<<blocks, 256, 0, stream>>>(W, K, N, Npad, dst);
    return dst;
  };
  bf16_t* Win_t = wt(Win, EMBW, DM, DM);
  bf16_t *Wq_t[6], *Wk_t[6], *Wv_t[6], *Wo_t[6], *W1_t[6], *W2_t[6];
  for (int l = 0; l < 6; ++l) {
    Wq_t[l] = wt(L[l].Wq, DM, DM, DM);
    Wk_t[l] = wt(L[l].Wk, DM, DM, DM);
    Wv_t[l] = wt(L[l].Wv, DM, DM, DM);
    Wo_t[l] = wt(L[l].Wo, DM, DM, DM);
    W1_t[l] = wt(L[l].W1, DM, FF, FF);
    W2_t[l] = wt(L[l].W2, FF, DM, DM);
  }
  bf16_t* Wtype_t = wt(Wtype, DM, 8, 128);
  bf16_t* Wcat_t  = wt(Wcat, 288, DM, DM);
  bf16_t* Wtem_t  = wt(Wp_tempo,    288, 64,  128);
  bf16_t* Wcho_t  = wt(Wp_chord,    288, 256, 256);
  bf16_t* Wbar_t  = wt(Wp_barbeat,  288, 32,  128);
  bf16_t* Wpit_t  = wt(Wp_pitch,    288, 128, 128);
  bf16_t* Wdur_t  = wt(Wp_duration, 288, 64,  128);
  bf16_t* Wvel_t  = wt(Wp_velocity, 288, 64,  128);

  // activations
  float*  h    = (float*) alloc((size_t)MTOT * DM * 4);
  bf16_t* hb   = (bf16_t*)alloc((size_t)MTOT * DM * 2);
  bf16_t* qb   = (bf16_t*)alloc((size_t)MTOT * DM * 2);
  bf16_t* kb   = (bf16_t*)alloc((size_t)MTOT * DM * 2);
  bf16_t* vb   = (bf16_t*)alloc((size_t)MTOT * DM * 2);
  bf16_t* ab   = (bf16_t*)alloc((size_t)MTOT * DM * 2);
  bf16_t* embc = (bf16_t*)alloc((size_t)MC * EMBW * 2);
  bf16_t* f1   = (bf16_t*)alloc((size_t)MC * FF * 2);
  bf16_t* A1   = (bf16_t*)alloc((size_t)MTOT * 288 * 2);
  bf16_t* catb = (bf16_t*)alloc((size_t)MTOT * 288 * 2);
  float*  yf   = (float*) alloc((size_t)MTOT * DM * 4);

  auto gemm = [&](const bf16_t* A, int lda, const bf16_t* Bt, int ldb,
                  const float* bias, const float* resid, int ldr,
                  float* Cf, int ldc, bf16_t* Cb, int ldcb,
                  int M, int N, int K, int flags) {
    dim3 grid(M / 256, (N + 127) / 128);
    gemm_bf16_kernel<<<grid, 256, 0, stream>>>(A, lda, Bt, ldb, bias, resid, ldr,
                                               Cf, ldc, Cb, ldcb, M, N, K, flags);
  };

  // ---- input: embeddings -> Win -> +pe ----
  for (int c = 0; c < MTOT / MC; ++c) {
    int m0 = c * MC;
    long long tot = (long long)MC * EMBW;
    embed_kernel<<<(int)((tot + 255) / 256), 256, 0, stream>>>(
        x, m0, MC, emb[0], emb[1], emb[2], emb[3], emb[4], emb[5], emb[6], embc);
    gemm(embc, EMBW, Win_t, EMBW, bin, nullptr, 0, h + (size_t)m0 * DM, DM,
         nullptr, 0, MC, DM, EMBW, 0);
  }
  {
    long long tot = (long long)MTOT * DM;
    add_pe_kernel<<<(int)((tot + 255) / 256), 256, 0, stream>>>(h, hb, pe);
  }

  // ---- encoder layers ----
  for (int l = 0; l < 6; ++l) {
    gemm(hb, DM, Wq_t[l], DM, L[l].bq, nullptr, 0, nullptr, 0, qb, DM, MTOT, DM, DM, 0);
    gemm(hb, DM, Wk_t[l], DM, L[l].bk, nullptr, 0, nullptr, 0, kb, DM, MTOT, DM, DM, 0);
    gemm(hb, DM, Wv_t[l], DM, L[l].bv, nullptr, 0, nullptr, 0, vb, DM, MTOT, DM, DM, 0);
    attn_scan_kernel<<<BB * 4, 64, 0, stream>>>(qb, kb, vb, ab);
    gemm(ab, DM, Wo_t[l], DM, L[l].bo, h, DM, h, DM, nullptr, 0, MTOT, DM, DM, 0);
    ln_kernel<<<MTOT, 256, 0, stream>>>(h, hb, L[l].g1, L[l].b1);
    for (int c = 0; c < MTOT / MC; ++c) {
      int m0 = c * MC;
      gemm(hb + (size_t)m0 * DM, DM, W1_t[l], DM, L[l].c1, nullptr, 0,
           nullptr, 0, f1, FF, MC, FF, DM, /*gelu*/1);
      gemm(f1, FF, W2_t[l], FF, L[l].c2, h + (size_t)m0 * DM, DM,
           h + (size_t)m0 * DM, DM, nullptr, 0, MC, DM, FF, 0);
    }
    ln_kernel<<<MTOT, 256, 0, stream>>>(h, hb, L[l].g2, L[l].b2);
  }
  // final encoder LayerNorm
  ln_kernel<<<MTOT, 256, 0, stream>>>(h, hb, gf, bfin);

  // ---- output heads ----
  float* out = (float*)d_out;
  float* o_tempo   = out;
  float* o_chord   = o_tempo   + (size_t)MTOT * 64;
  float* o_barbeat = o_chord   + (size_t)MTOT * 256;
  float* o_type    = o_barbeat + (size_t)MTOT * 32;
  float* o_pitch   = o_type    + (size_t)MTOT * 8;
  float* o_dur     = o_pitch   + (size_t)MTOT * 128;
  float* o_vel     = o_dur     + (size_t)MTOT * 64;
  float* o_cls     = o_vel     + (size_t)MTOT * 64;

  gemm(hb, DM, Wtype_t, DM, btype, nullptr, 0, o_type, 8, nullptr, 0, MTOT, 8, DM, 0);

  {
    long long tot = (long long)MTOT * 288;
    build_a1_kernel<<<(int)((tot + 255) / 256), 256, 0, stream>>>(hb, target, emb[3], A1);
  }
  // y_ = [h|tf] @ Wcat + bcat ; f32 into yf, bf16 into catb[:, :256]
  gemm(A1, 288, Wcat_t, 288, bcat, nullptr, 0, yf, DM, catb, 288, MTOT, DM, 288, 0);
  {
    long long tot = (long long)MTOT * 32;
    build_label_kernel<<<(int)((tot + 255) / 256), 256, 0, stream>>>(label, emb_label, catb);
  }
  gemm(catb, 288, Wtem_t, 288, bp_tempo,    nullptr, 0, o_tempo,   64,  nullptr, 0, MTOT, 64,  288, 0);
  gemm(catb, 288, Wcho_t, 288, bp_chord,    nullptr, 0, o_chord,   256, nullptr, 0, MTOT, 256, 288, 0);
  gemm(catb, 288, Wbar_t, 288, bp_barbeat,  nullptr, 0, o_barbeat, 32,  nullptr, 0, MTOT, 32,  288, 0);
  gemm(catb, 288, Wpit_t, 288, bp_pitch,    nullptr, 0, o_pitch,   128, nullptr, 0, MTOT, 128, 288, 0);
  gemm(catb, 288, Wdur_t, 288, bp_duration, nullptr, 0, o_dur,     64,  nullptr, 0, MTOT, 64,  288, 0);
  gemm(catb, 288, Wvel_t, 288, bp_velocity, nullptr, 0, o_vel,     64,  nullptr, 0, MTOT, 64,  288, 0);

  cls_kernel<<<BB, 256, 0, stream>>>(yf, Wcls, bcls, o_cls);
}
